// RNN_2817498546846
// MI455X (gfx1250) — matrix operand order repaired and checked by reading, not compiled, not run on hardware
//
#include <hip/hip_runtime.h>
#include <hip/hip_bf16.h>

#define BB 512
#define FF 64
#define TT 64
#define HH 512
#define KK 4
#define H3 1536

typedef __attribute__((ext_vector_type(16))) _Float16 v16h;
typedef __attribute__((ext_vector_type(8)))  _Float16 v8h;
typedef __attribute__((ext_vector_type(8)))  float    v8f;
typedef unsigned int u32;
typedef u32 u32x4 __attribute__((ext_vector_type(4)));
typedef int i32x4v __attribute__((ext_vector_type(4)));
typedef int i32x8v __attribute__((ext_vector_type(8)));

// ---- WMMA fragment loaders (wave32, 16x16x32 f16) ----
// A-matrix: lane L holds row r = L%16; halves 0..15 = K[kb..kb+15],
// kb = (L<16) ? 0 : 16  -> two 16-byte global loads per fragment.
__device__ __forceinline__ v16h frag_a(const _Float16* __restrict__ p, int ld,
                                       int r0, int c0, int lane) {
  int r = r0 + (lane & 15);
  int c = c0 + ((lane & 16) ? 16 : 0);
  const _Float16* q = p + (size_t)r * ld + c;
  v8h lo = *(const v8h*)q;
  v8h hi = *(const v8h*)(q + 8);
  v16h out;
#pragma unroll
  for (int i = 0; i < 8; ++i) { out[i] = lo[i]; out[i + 8] = hi[i]; }
  return out;
}

// B-matrix (K x N = 32x16): lane L holds column n = L%16, K-halves 0..15 / 16..31 by L/16.
// B[k,n] = W[row n, col k] for out = X * W^T, so one contiguous 32-byte run per lane.
__device__ __forceinline__ v16h frag_b(const _Float16* __restrict__ p, int ld,
                                       int r0, int c0, int lane) {
  int r = r0 + (lane & 15);
  int c = c0 + ((lane & 16) ? 16 : 0);
  return *(const v16h*)(p + (size_t)r * ld + c);
}

// Same layout, but sourced from an LDS-staged 16x32 tile (row-major, ld=32).
__device__ __forceinline__ v16h frag_b_lds(const _Float16* tile, int lane) {
  int r = lane & 15;
  int c = (lane & 16) ? 16 : 0;
  return *(const v16h*)(tile + r * 32 + c);
}

#define WMMA(a, b, c) __builtin_amdgcn_wmma_f32_16x16x32_f16(false, (a), false, (b), (short)0, (c), false, false)

// ---- TDM: stage one 16-row x 32-half tile (row stride 512 halves) into LDS ----
// D# per CDNA5 ISA: 2D descriptor, data_size=2B,
// tile_dim0=32, tile_dim1=16, tensor_dim0_stride=512, groups 2/3 zero.
__device__ __forceinline__ void tdm_load_tile(const _Float16* gptr, u32 lds_off) {
  unsigned long long ga = (unsigned long long)(uintptr_t)gptr;
  u32x4 g0;
  g0[0] = 1u;                                                // count=1, user mode
  g0[1] = lds_off;                                           // LDS byte address
  g0[2] = (u32)ga;                                           // global_addr[31:0]
  g0[3] = (u32)((ga >> 32) & 0x01FFFFFFu) | (2u << 30);      // global_addr[56:32], type=2
  i32x8v g1;
  g1[0] = (int)(1u << 16);                                   // wg_mask=0, data_size=1 (2B)
  g1[1] = (int)((512u & 0xFFFFu) << 16);                     // tensor_dim0[15:0] @ bit48
  g1[2] = (int)((512u >> 16) | ((1048576u & 0xFFFFu) << 16));// tdim0 hi, tensor_dim1 lo
  g1[3] = (int)((1048576u >> 16) | (32u << 16));             // tdim1 hi, tile_dim0=32
  g1[4] = 16;                                                // tile_dim1=16, tile_dim2=0
  g1[5] = 512;                                               // tensor_dim0_stride lo32
  g1[6] = 0;                                                 // stride0 hi16, stride1 lo16
  g1[7] = 0;                                                 // stride1 hi32
  i32x4v z4 = {};
#if defined(__clang_major__) && (__clang_major__ >= 23)
  i32x8v z8 = {};
  __builtin_amdgcn_tensor_load_to_lds(g0, g1, z4, z4, z8, 0);
#else
  __builtin_amdgcn_tensor_load_to_lds(g0, g1, z4, z4, 0);
#endif
}

// ---- fp32 -> f16 conversion (weights) ----
__global__ void k_cvt_f16(_Float16* __restrict__ dst, const float* __restrict__ src, int n) {
  int i = blockIdx.x * blockDim.x + threadIdx.x;
  int stride = gridDim.x * blockDim.x;
  for (; i < n; i += stride) dst[i] = (_Float16)src[i];
}

// ---- data[b,f,t] -> dataT[t,b,f] (f16) ----
__global__ void k_transpose(_Float16* __restrict__ dT, const float* __restrict__ d) {
  const int n = BB * FF * TT;
  int i = blockIdx.x * blockDim.x + threadIdx.x;
  int stride = gridDim.x * blockDim.x;
  for (; i < n; i += stride) {
    int t = i % TT;
    int f = (i / TT) % FF;
    int b = i / (TT * FF);
    dT[(size_t)t * BB * FF + (size_t)b * FF + f] = (_Float16)d[i];
  }
}

// ---- broadcast init -> h0 (f32 + f16), zero the scalar-output accumulator ----
__global__ void k_init(float* __restrict__ h32, _Float16* __restrict__ h16,
                       float* __restrict__ sbuf, const float* __restrict__ init) {
  const int n = KK * BB * HH;
  int i = blockIdx.x * blockDim.x + threadIdx.x;
  int stride = gridDim.x * blockDim.x;
  for (int j = i; j < n; j += stride) {
    float v = init[j % (BB * HH)];
    h32[j] = v;
    h16[j] = (_Float16)v;
  }
  for (int j = i; j < TT * KK * BB; j += stride) sbuf[j] = 0.f;
}

// ---- input projection: xseq[t',k,b,h] = relu(data[b,:,t] @ Win[k]^T + b[k]) ----
__global__ void __launch_bounds__(256) k_proj(const _Float16* __restrict__ dT,
                                              const _Float16* __restrict__ hWin,
                                              const float* __restrict__ fcb,
                                              _Float16* __restrict__ xseq) {
  int lane = threadIdx.x & 31;
  int w = blockIdx.x * 8 + (threadIdx.x >> 5);   // 8192 waves: (t,k,m)
  int m = w & 31;
  int rem = w >> 5;
  int k = rem & 3;
  int t = rem >> 2;
  const _Float16* A = dT + (size_t)t * BB * FF;
  const _Float16* W = hWin + (size_t)k * HH * FF;
  v16h a0 = frag_a(A, FF, m * 16, 0, lane);
  v16h a1 = frag_a(A, FF, m * 16, 32, lane);
  int hc = lane & 15;
  int mAdd = (lane & 16) ? 8 : 0;
  int tst = (k < 2) ? t : (TT - 1 - t);
  _Float16* xo = xseq + ((size_t)tst * KK + k) * BB * HH;
  for (int n = 0; n < 32; ++n) {
    v16h b0 = frag_b(W, FF, n * 16, 0, lane);
    v16h b1 = frag_b(W, FF, n * 16, 32, lane);
    v8f acc = {};
    acc = WMMA(a0, b0, acc);
    acc = WMMA(a1, b1, acc);
    int hcol = n * 16 + hc;
    float bias = fcb[k * HH + hcol];
#pragma unroll
    for (int j = 0; j < 8; ++j) {
      float v = acc[j] + bias;
      v = v > 0.f ? v : 0.f;
      int mg = m * 16 + j + mAdd;
      xo[(size_t)mg * HH + hcol] = (_Float16)v;
    }
  }
}

// ---- one GRU timestep: TDM-staged weights in LDS + 6 WMMA accumulators + gates ----
// Block = 8 waves = (k, n, m-group of 8). The 6 weight fragments per k-step are
// identical across all 8 waves, so wave 0 streams them through the Tensor Data
// Mover into a double-buffered LDS stage (s_wait_tensorcnt + barrier pipeline).
__global__ void __launch_bounds__(256) k_gru_step(
    int t,
    const _Float16* __restrict__ xseq, const _Float16* __restrict__ hWih,
    const _Float16* __restrict__ hWhh,
    const float* __restrict__ bih, const float* __restrict__ bhh,
    const float* __restrict__ h32c, const _Float16* __restrict__ h16c,
    float* __restrict__ h32n, _Float16* __restrict__ h16n,
    const float* __restrict__ wout, float* __restrict__ sbuf) {
  __shared__ _Float16 smem[2 * 6 * 512];   // 2 buffers x 6 tiles x (16x32) halves = 12 KB

  int lane = threadIdx.x & 31;
  int wave = threadIdx.x >> 5;
  int mg = blockIdx.x & 3;                 // m-group
  int n = (blockIdx.x >> 2) & 31;
  int k = blockIdx.x >> 7;
  int m = mg * 8 + wave;

  const _Float16* X  = xseq + ((size_t)t * KK + k) * BB * HH;
  const _Float16* Hc = h16c + (size_t)k * BB * HH;
  const _Float16* Wi = hWih + (size_t)k * H3 * HH;
  const _Float16* Wh = hWhh + (size_t)k * H3 * HH;

  // row bases for the six weight tiles (r,z,n gates of Wih and Whh)
  const _Float16* wrow[6] = {
      Wi + (size_t)(n * 16) * HH,            Wi + (size_t)(HH + n * 16) * HH,
      Wi + (size_t)(2 * HH + n * 16) * HH,   Wh + (size_t)(n * 16) * HH,
      Wh + (size_t)(HH + n * 16) * HH,       Wh + (size_t)(2 * HH + n * 16) * HH };

  // prologue: stage k-step 0 into buffer 0
  if (wave == 0) {
#pragma unroll
    for (int i = 0; i < 6; ++i)
      tdm_load_tile(wrow[i], (u32)(i * 512 * 2));
  }

  v8f cir = {}, ciz = {}, cin = {}, chr_ = {}, chz = {}, chn = {};
  for (int kk = 0; kk < 16; ++kk) {
    int buf = kk & 1;
    if (wave == 0) __builtin_amdgcn_s_wait_tensorcnt(0);
    __syncthreads();                       // buf ready; prior readers of buf^1 done
    if (wave == 0 && kk < 15) {            // overlap TDM for kk+1 with this k-step's math
      int c1 = (kk + 1) * 32;
#pragma unroll
      for (int i = 0; i < 6; ++i)
        tdm_load_tile(wrow[i] + c1, (u32)(((buf ^ 1) * 6 + i) * 512 * 2));
    }
    int c0 = kk * 32;
    v16h ax  = frag_a(X,  HH, m * 16, c0, lane);
    v16h ah  = frag_a(Hc, HH, m * 16, c0, lane);
    const _Float16* sb0 = &smem[buf * 6 * 512];
    v16h bir = frag_b_lds(sb0 + 0 * 512, lane);
    v16h biz = frag_b_lds(sb0 + 1 * 512, lane);
    v16h bin = frag_b_lds(sb0 + 2 * 512, lane);
    v16h bhr = frag_b_lds(sb0 + 3 * 512, lane);
    v16h bhz = frag_b_lds(sb0 + 4 * 512, lane);
    v16h bhn = frag_b_lds(sb0 + 5 * 512, lane);
    cir  = WMMA(ax, bir, cir);
    ciz  = WMMA(ax, biz, ciz);
    cin  = WMMA(ax, bin, cin);
    chr_ = WMMA(ah, bhr, chr_);
    chz  = WMMA(ah, bhz, chz);
    chn  = WMMA(ah, bhn, chn);
  }

  // epilogue: gates in f32, write new state (f32 + f16), fused output-head partial dot
  int hc = n * 16 + (lane & 15);
  int mAdd = (lane & 16) ? 8 : 0;
  float b_ir = bih[k * H3 + hc], b_iz = bih[k * H3 + HH + hc], b_in = bih[k * H3 + 2 * HH + hc];
  float b_hr = bhh[k * H3 + hc], b_hz = bhh[k * H3 + HH + hc], b_hn = bhh[k * H3 + 2 * HH + hc];
  float wo = wout[(k & 1) * HH + hc];
  const float* h32k = h32c + (size_t)k * BB * HH;
  float* h32o = h32n + (size_t)k * BB * HH;
  _Float16* h16o = h16n + (size_t)k * BB * HH;
  float* sb = sbuf + ((size_t)t * KK + k) * BB;
#pragma unroll
  for (int j = 0; j < 8; ++j) {
    int mgl = m * 16 + j + mAdd;
    size_t idx = (size_t)mgl * HH + hc;
    float hold = h32k[idx];
    float r = 1.f / (1.f + __expf(-(cir[j] + b_ir + chr_[j] + b_hr)));
    float z = 1.f / (1.f + __expf(-(ciz[j] + b_iz + chz[j] + b_hz)));
    float npre = cin[j] + b_in + r * (chn[j] + b_hn);
    float e = __expf(-2.f * npre);
    float nn = (1.f - e) / (1.f + e);
    float hnew = (1.f - z) * nn + z * hold;
    h32o[idx] = hnew;
    h16o[idx] = (_Float16)hnew;
    float pv = hnew * wo;                    // reduce over the 16 h-columns (one half-wave)
    pv += __shfl_xor(pv, 1, 32);
    pv += __shfl_xor(pv, 2, 32);
    pv += __shfl_xor(pv, 4, 32);
    pv += __shfl_xor(pv, 8, 32);
    if ((lane & 15) == 0) atomicAdd(&sb[mgl], pv);
  }
}

// ---- combine fwd/bwd scalar streams into [air(B,T), bed(B,T)] ----
__global__ void k_final(const float* __restrict__ sbuf, const float* __restrict__ fob,
                        float* __restrict__ out) {
  int i = blockIdx.x * blockDim.x + threadIdx.x;
  if (i >= BB * TT) return;
  int b = i / TT, t = i % TT;
  out[(size_t)b * TT + t] =
      sbuf[(size_t)t * KK * BB + 0 * BB + b] + sbuf[(size_t)(TT - 1 - t) * KK * BB + 2 * BB + b] + fob[0];
  out[(size_t)BB * TT + (size_t)b * TT + t] =
      sbuf[(size_t)t * KK * BB + 1 * BB + b] + sbuf[(size_t)(TT - 1 - t) * KK * BB + 3 * BB + b] + fob[1];
}

extern "C" void kernel_launch(void* const* d_in, const int* in_sizes, int n_in,
                              void* d_out, int out_size, void* d_ws, size_t ws_size,
                              hipStream_t stream) {
  const float* data    = (const float*)d_in[0];
  const float* initv   = (const float*)d_in[1];
  const float* fc_in_W = (const float*)d_in[2];
  const float* fc_in_b = (const float*)d_in[3];
  const float* Wih     = (const float*)d_in[4];
  const float* Whh     = (const float*)d_in[5];
  const float* bih     = (const float*)d_in[6];
  const float* bhh     = (const float*)d_in[7];
  const float* fc_out_W = (const float*)d_in[8];
  const float* fc_out_b = (const float*)d_in[9];
  float* out = (float*)d_out;

  char* ws = (char*)d_ws;
  size_t off = 0;
  _Float16* hWih = (_Float16*)(ws + off); off += (size_t)KK * H3 * HH * 2;
  _Float16* hWhh = (_Float16*)(ws + off); off += (size_t)KK * H3 * HH * 2;
  _Float16* hWin = (_Float16*)(ws + off); off += (size_t)KK * HH * FF * 2;
  _Float16* dataT = (_Float16*)(ws + off); off += (size_t)TT * BB * FF * 2;
  _Float16* xseq = (_Float16*)(ws + off); off += (size_t)TT * KK * BB * HH * 2;
  float* h32a = (float*)(ws + off); off += (size_t)KK * BB * HH * 4;
  float* h32b = (float*)(ws + off); off += (size_t)KK * BB * HH * 4;
  _Float16* h16a = (_Float16*)(ws + off); off += (size_t)KK * BB * HH * 2;
  _Float16* h16b = (_Float16*)(ws + off); off += (size_t)KK * BB * HH * 2;
  float* sbuf = (float*)(ws + off); off += (size_t)TT * KK * BB * 4;

  // 1) precision conversion of the static weights
  k_cvt_f16<<<1024, 256, 0, stream>>>(hWih, Wih, KK * H3 * HH);
  k_cvt_f16<<<1024, 256, 0, stream>>>(hWhh, Whh, KK * H3 * HH);
  k_cvt_f16<<<256, 256, 0, stream>>>(hWin, fc_in_W, KK * HH * FF);

  // 2) data transpose to GEMM-friendly [t,b,f] f16
  k_transpose<<<2048, 256, 0, stream>>>(dataT, data);

  // 3) broadcast initial hidden state, zero the head accumulator
  k_init<<<2048, 256, 0, stream>>>(h32a, h16a, sbuf, initv);

  // 4) input projection (WMMA), reversal folded into the store
  k_proj<<<1024, 256, 0, stream>>>(dataT, hWin, fc_in_b, xseq);

  // 5) 64 recurrent steps, double-buffered state
  float* hc32 = h32a; float* hn32 = h32b;
  _Float16* hc16 = h16a; _Float16* hn16 = h16b;
  for (int t = 0; t < TT; ++t) {
    k_gru_step<<<512, 256, 0, stream>>>(t, xseq, hWih, hWhh, bih, bhh,
                                        hc32, hc16, hn32, hn16, fc_out_W, sbuf);
    float* t32 = hc32; hc32 = hn32; hn32 = t32;
    _Float16* t16 = hc16; hc16 = hn16; hn16 = t16;
  }

  // 6) combine forward/backward streams + output bias
  k_final<<<(BB * TT + 255) / 256, 256, 0, stream>>>(sbuf, fc_out_b, out);
}
